// MultiHeadedAttention_64441689309679
// MI455X (gfx1250) — matrix operand order repaired and checked by reading, not compiled, not run on hardware
//
#include <hip/hip_runtime.h>
#include <hip/hip_bf16.h>

// MI455X / gfx1250, wave32. All matmuls via v_wmma_f32_16x16x32_f16.
#define S_LEN 2048
#define D_DIM 1024
#define H_NUM 16
#define DKD   64
#define B_NUM 2
#define M_ROWS (B_NUM * S_LEN)   // 4096

typedef __attribute__((ext_vector_type(16))) _Float16 v16h;
typedef __attribute__((ext_vector_type(8)))  _Float16 v8h;
typedef __attribute__((ext_vector_type(4)))  _Float16 v4h;
typedef __attribute__((ext_vector_type(8)))  float    v8f;
typedef __attribute__((ext_vector_type(4)))  float    v4f;

static __device__ __forceinline__ v8f wmma16(v16h a, v16h b, v8f c) {
  // (neg_a, A, neg_b, B, c_mod, C, reuse_a, reuse_b) -> v_wmma_f32_16x16x32_f16
  return __builtin_amdgcn_wmma_f32_16x16x32_f16(false, a, false, b, (short)0, c,
                                                false, false);
}
static __device__ __forceinline__ v16h cat8(v8h lo, v8h hi) {
  return __builtin_shufflevector(lo, hi, 0, 1, 2, 3, 4, 5, 6, 7,
                                          8, 9, 10, 11, 12, 13, 14, 15);
}

// ---------------------------------------------------------------------------
// Tensor Data Mover: 1-D contiguous f16 copy global -> LDS.
// D# per CDNA5 ISA (§8.3/8.4): count=1, type=2,
// data_size=2B, tile_dim0 = nelem, tensor_dim0 = nelem, tile_dim1 = 1.
// ---------------------------------------------------------------------------
#if defined(__AMDGCN__) && __has_builtin(__builtin_amdgcn_tensor_load_to_lds)
#define USE_TDM 1
#else
#define USE_TDM 0
#endif

#if USE_TDM
typedef __attribute__((ext_vector_type(4))) unsigned int u32x4;
typedef __attribute__((ext_vector_type(8))) int          i32x8;
typedef __attribute__((ext_vector_type(4))) int          i32x4;

static __device__ __forceinline__ void tdm_copy_f16(const _Float16* gsrc,
                                                    void* lds_dst,
                                                    unsigned nelem) {
  const unsigned long long ga = (unsigned long long)(size_t)gsrc;
  // Flat LDS pointers carry the LDS byte offset in the low 32 bits.
  const unsigned lds_addr = (unsigned)(size_t)lds_dst;
  u32x4 g0;
  g0[0] = 1u;                                     // count=1, user mode
  g0[1] = lds_addr;                               // lds_addr (bytes)
  g0[2] = (unsigned)(ga & 0xffffffffull);         // global_addr[31:0]
  g0[3] = (unsigned)((ga >> 32) & 0x01ffffffull)  // global_addr[56:32]
          | (2u << 30);                           // type = 2 ("image")
  i32x8 g1;
  g1[0] = (int)(1u << 16);                        // data_size = 1 -> 2 bytes
  g1[1] = (int)((nelem & 0xffffu) << 16);         // tensor_dim0[15:0]
  g1[2] = (int)((nelem >> 16) & 0xffffu)          // tensor_dim0[31:16]
          | (int)(1u << 16);                      // tensor_dim1 = 1
  g1[3] = (int)(nelem << 16);                     // tile_dim0 = nelem (16b)
  g1[4] = 1;                                      // tile_dim1 = 1, tile_dim2 = 0
  g1[5] = (int)nelem;                             // tensor_dim0_stride lo
  g1[6] = 0;
  g1[7] = 0;
  i32x4 z4 = {0, 0, 0, 0};
#if defined(__clang_major__) && (__clang_major__ >= 23)
  i32x8 z8 = {0, 0, 0, 0, 0, 0, 0, 0};
  __builtin_amdgcn_tensor_load_to_lds(g0, g1, z4, z4, z8, 0);
#else
  __builtin_amdgcn_tensor_load_to_lds(g0, g1, z4, z4, 0);
#endif
}
#endif

// ---------------------------------------------------------------------------
// GEMM: Y[M,N] = (X[M,K] @ W[N,K]^T + bias[N]) * out_scale
//   A_IS_F16 = false : X is f32 (input activations), converted to f16 in LDS
//   A_IS_F16 = true  : X is f16 (attention output in workspace)
//   HEAD_OUT = true  : Y stored f16 as [B, H, S, DK] (head-major for attention)
//   HEAD_OUT = false : Y stored f32 as [M, N] (final kernel output)
// Block: 128 threads = 4 waves; tile 64(M) x 64(N); K-step 32.
// ---------------------------------------------------------------------------
template <bool A_IS_F16, bool HEAD_OUT>
__global__ __launch_bounds__(128) void gemm_xwt(const void* __restrict__ Ain,
                                                const float* __restrict__ W,
                                                const float* __restrict__ bias,
                                                void* __restrict__ Yout,
                                                float out_scale) {
  __shared__ _Float16 As[64][32];
  __shared__ _Float16 Bs[64][32];

  const int nb   = blockIdx.x * 64;
  const int mb   = blockIdx.y * 64;
  const int tid  = threadIdx.x;
  const int wave = tid >> 5;
  const int lane = tid & 31;
  const int ln   = lane & 15;   // C-layout: N within tile / A-layout: M
  const int hs   = lane >> 4;   // half-wave select

  v8f acc[4];
#pragma unroll
  for (int t = 0; t < 4; ++t) acc[t] = (v8f){0.f, 0.f, 0.f, 0.f, 0.f, 0.f, 0.f, 0.f};

  const int sr = tid >> 1;        // staging row 0..63
  const int sc = (tid & 1) * 16;  // staging col 0 / 16

  for (int kb = 0; kb < D_DIM; kb += 32) {
    // ---- stage A tile (64 x 32) into LDS as f16 ----
    if (A_IS_F16) {
      const _Float16* ap = (const _Float16*)Ain + (size_t)(mb + sr) * D_DIM + kb + sc;
      *(v8h*)&As[sr][sc]     = *(const v8h*)ap;
      *(v8h*)&As[sr][sc + 8] = *(const v8h*)(ap + 8);
    } else {
      const float* ap = (const float*)Ain + (size_t)(mb + sr) * D_DIM + kb + sc;
#pragma unroll
      for (int j = 0; j < 16; j += 4) {
        v4f x = *(const v4f*)(ap + j);
        v4h y;
        y[0] = (_Float16)x[0]; y[1] = (_Float16)x[1];
        y[2] = (_Float16)x[2]; y[3] = (_Float16)x[3];
        *(v4h*)&As[sr][sc + j] = y;
      }
    }
    // ---- stage B tile: rows of W (x@W^T => B[k][n] = W[n][k]) ----
    {
      const float* wp = W + (size_t)(nb + sr) * D_DIM + kb + sc;
#pragma unroll
      for (int j = 0; j < 16; j += 4) {
        v4f x = *(const v4f*)(wp + j);
        v4h y;
        y[0] = (_Float16)x[0]; y[1] = (_Float16)x[1];
        y[2] = (_Float16)x[2]; y[3] = (_Float16)x[3];
        *(v4h*)&Bs[sr][sc + j] = y;
      }
    }
    __syncthreads();

    // A fragment: lane holds row (wave*16 + ln); K chunks {hs*8..+7, 16+hs*8..+7}
    v8h alo = *(const v8h*)&As[wave * 16 + ln][hs * 8];
    v8h ahi = *(const v8h*)&As[wave * 16 + ln][16 + hs * 8];
    v16h af = cat8(alo, ahi);
#pragma unroll
    for (int t = 0; t < 4; ++t) {
      // B fragment: lane holds column n = t*16+ln; K chunks {hs*8..+7, 16+hs*8..+7}
      v8h b0 = *(const v8h*)&Bs[t * 16 + ln][hs * 8];
      v8h b1 = *(const v8h*)&Bs[t * 16 + ln][16 + hs * 8];
      acc[t] = wmma16(af, cat8(b0, b1), acc[t]);
    }
    __syncthreads();
  }

  // ---- epilogue: bias, scale, store ----
#pragma unroll
  for (int t = 0; t < 4; ++t) {
    const int n    = nb + t * 16 + ln;
    const float bv = bias[n];
#pragma unroll
    for (int r = 0; r < 8; ++r) {
      const int m   = mb + wave * 16 + r + hs * 8;  // C layout: M = r + 8*hs
      const float y = (acc[t][r] + bv) * out_scale;
      if (HEAD_OUT) {
        const int bb = m >> 11, ss = m & (S_LEN - 1);
        const int hh = n >> 6, dd = n & 63;
        ((_Float16*)Yout)[(((size_t)(bb * H_NUM + hh) * S_LEN) + ss) * DKD + dd] =
            (_Float16)y;
      } else {
        ((float*)Yout)[(size_t)m * D_DIM + n] = y;
      }
    }
  }
}

// ---------------------------------------------------------------------------
// Flash attention (causal). Q/K/V f16 in [B,H,S,DK]; O f16 in [B,S,D].
// Block: 128 threads = 4 waves; block owns 64 q rows; wave owns 16.
// kv tiles of 64 keys: K/V fetched via Tensor Data Mover (quarter tile per
// wave), V transposed LDS->LDS for contiguous P@V B-fragments.
// Row sums computed by WMMA against an all-ones B matrix (replaces the
// ds_bpermute butterfly); row max still uses a 16-lane xor butterfly.
// ---------------------------------------------------------------------------
__global__ __launch_bounds__(128) void attn_kernel(const _Float16* __restrict__ Q,
                                                   const _Float16* __restrict__ K,
                                                   const _Float16* __restrict__ V,
                                                   _Float16* __restrict__ O) {
  __shared__ _Float16 Kt[64][64];     // [kv][dk]
  __shared__ _Float16 Vr[64][64];     // TDM landing: [kv][dk]
  __shared__ _Float16 Vt[64][64];     // transposed: [dk][kv]
  __shared__ _Float16 Pl[4][16][64];  // per-wave P tile (C->A relayout)

  const int qb0  = blockIdx.x * 64;
  const int h    = blockIdx.y;
  const int b    = blockIdx.z;
  const int tid  = threadIdx.x;
  const int wave = tid >> 5;
  const int lane = tid & 31;
  const int ln   = lane & 15;
  const int hs   = lane >> 4;

  const size_t headoff = (size_t)(b * H_NUM + h) * S_LEN * DKD;
  const _Float16* Qh = Q + headoff;
  const _Float16* Kh = K + headoff;
  const _Float16* Vh = V + headoff;

  const int wq = qb0 + wave * 16;  // this wave's first q row

  // Q fragments held in registers (softmax scale already folded into Q)
  v16h qf[2];
  {
    const _Float16* qrow = Qh + (size_t)(wq + ln) * DKD;
#pragma unroll
    for (int g = 0; g < 2; ++g) {
      v8h lo = *(const v8h*)(qrow + g * 32 + hs * 8);
      v8h hi = *(const v8h*)(qrow + g * 32 + hs * 8 + 16);
      qf[g]  = cat8(lo, hi);
    }
  }

  // all-ones B fragment: P @ ones gives row sums replicated in C layout
  v16h ones;
#pragma unroll
  for (int i = 0; i < 16; ++i) ones[i] = (_Float16)1.0f;

  v8f o[4];
#pragma unroll
  for (int t = 0; t < 4; ++t) o[t] = (v8f){0.f, 0.f, 0.f, 0.f, 0.f, 0.f, 0.f, 0.f};
  float mrow[8], lrow[8];
#pragma unroll
  for (int r = 0; r < 8; ++r) { mrow[r] = -1e30f; lrow[r] = 0.f; }

  const int sr = tid >> 1;            // staging kv row
  const int sc = (tid & 1) * 32;      // staging dk col 0/32
  const int ntiles = (qb0 >> 6) + 1;  // causal: kv_base <= qb0

  for (int kt = 0; kt < ntiles; ++kt) {
    const int kvb = kt * 64;

    // ---- fetch K and V tiles (quarter tile per wave) ----
#if USE_TDM
    tdm_copy_f16(Kh + (size_t)(kvb + wave * 16) * DKD, &Kt[wave * 16][0], 16 * DKD);
    tdm_copy_f16(Vh + (size_t)(kvb + wave * 16) * DKD, &Vr[wave * 16][0], 16 * DKD);
    __builtin_amdgcn_s_wait_tensorcnt(0);
#else
    {
      const _Float16* kp = Kh + (size_t)(kvb + sr) * DKD + sc;
      const _Float16* vp = Vh + (size_t)(kvb + sr) * DKD + sc;
#pragma unroll
      for (int j = 0; j < 32; j += 8) {
        *(v8h*)&Kt[sr][sc + j] = *(const v8h*)(kp + j);
        *(v8h*)&Vr[sr][sc + j] = *(const v8h*)(vp + j);
      }
    }
#endif
    __syncthreads();

    // ---- prefetch next kv tile into GL2 while we compute ----
    if (kt + 1 < ntiles) {
      const _Float16* np =
          (tid < 64 ? Kh : Vh) + (size_t)(kvb + 64 + (tid & 63)) * DKD;
      __builtin_prefetch(np, 0, 1);
    }

    // ---- transpose V: Vr[kv][dk] -> Vt[dk][kv] ----
#pragma unroll
    for (int j = 0; j < 32; j += 8) {
      v8h x = *(const v8h*)&Vr[sr][sc + j];
#pragma unroll
      for (int i = 0; i < 8; ++i) Vt[sc + j + i][sr] = x[i];
    }

    // ---- S = (Q*scale) @ K^T : 16 x 64 per wave ----
    v8f s[4];
#pragma unroll
    for (int t = 0; t < 4; ++t) s[t] = (v8f){0.f, 0.f, 0.f, 0.f, 0.f, 0.f, 0.f, 0.f};
#pragma unroll
    for (int g = 0; g < 2; ++g) {
#pragma unroll
      for (int t = 0; t < 4; ++t) {
        // B[k][n] = K[kv = n][dk = k]: lane reads K row (t*16+ln), dk contiguous
        v8h b0 = *(const v8h*)&Kt[t * 16 + ln][g * 32 + hs * 8];
        v8h b1 = *(const v8h*)&Kt[t * 16 + ln][g * 32 + hs * 8 + 16];
        s[t]   = wmma16(qf[g], cat8(b0, b1), s[t]);
      }
    }
    // ---- causal mask: only diagonal tile needs per-element masking ----
    if (kvb == qb0) {
#pragma unroll
      for (int t = 0; t < 4; ++t)
#pragma unroll
        for (int r = 0; r < 8; ++r) {
          const int kv = kvb + t * 16 + ln;
          const int q  = wq + r + hs * 8;
          if (kv > q) s[t][r] = -1e30f;
        }
    }
    // ---- online softmax: row max via 16-lane butterfly ----
    float alpha[8];
#pragma unroll
    for (int r = 0; r < 8; ++r) {
      float x = fmaxf(fmaxf(s[0][r], s[1][r]), fmaxf(s[2][r], s[3][r]));
#pragma unroll
      for (int msk = 1; msk < 16; msk <<= 1) x = fmaxf(x, __shfl_xor(x, msk, 32));
      const float mn = fmaxf(mrow[r], x);
      alpha[r] = __expf(mrow[r] - mn);
      mrow[r]  = mn;
    }
    // ---- P = exp(S - m) written straight to LDS (C -> A relayout) ----
#pragma unroll
    for (int t = 0; t < 4; ++t)
#pragma unroll
      for (int r = 0; r < 8; ++r)
        Pl[wave][r + hs * 8][t * 16 + ln] = (_Float16)__expf(s[t][r] - mrow[r]);
    __syncthreads();  // Vt + Pl visibility across lanes/waves

    // ---- reload P as A fragments ----
    v16h pf[2];
#pragma unroll
    for (int g = 0; g < 2; ++g) {
      v8h plo = *(const v8h*)&Pl[wave][ln][g * 32 + hs * 8];
      v8h phi = *(const v8h*)&Pl[wave][ln][g * 32 + hs * 8 + 16];
      pf[g]   = cat8(plo, phi);
    }
    // ---- row sums via WMMA against ones (replaces shuffle butterfly) ----
    v8f su = (v8f){0.f, 0.f, 0.f, 0.f, 0.f, 0.f, 0.f, 0.f};
    su = wmma16(pf[0], ones, su);
    su = wmma16(pf[1], ones, su);
#pragma unroll
    for (int r = 0; r < 8; ++r) {
      lrow[r] = lrow[r] * alpha[r] + su[r];
#pragma unroll
      for (int t = 0; t < 4; ++t) o[t][r] *= alpha[r];
    }
    // ---- O += P @ V ----
#pragma unroll
    for (int g = 0; g < 2; ++g) {
#pragma unroll
      for (int t = 0; t < 4; ++t) {
        // B[k][n] = V[kv = k][dk = n]: transposed Vt gives contiguous reads
        v8h v0 = *(const v8h*)&Vt[t * 16 + ln][g * 32 + hs * 8];
        v8h v1 = *(const v8h*)&Vt[t * 16 + ln][g * 32 + hs * 8 + 16];
        o[t]   = wmma16(pf[g], cat8(v0, v1), o[t]);
      }
    }
    __syncthreads();  // protect Kt/Vr/Vt before next tile's fetch
  }

  // ---- normalize and store O as [B, S, D] f16 ----
#pragma unroll
  for (int t = 0; t < 4; ++t)
#pragma unroll
    for (int r = 0; r < 8; ++r) {
      const int q = wq + r + hs * 8;
      const int d = h * DKD + t * 16 + ln;
      O[((size_t)b * S_LEN + q) * D_DIM + d] = (_Float16)(o[t][r] / lrow[r]);
    }
}

// ---------------------------------------------------------------------------
extern "C" void kernel_launch(void* const* d_in, const int* in_sizes, int n_in,
                              void* d_out, int out_size, void* d_ws, size_t ws_size,
                              hipStream_t stream) {
  const float* query = (const float*)d_in[0];
  const float* key   = (const float*)d_in[1];
  const float* value = (const float*)d_in[2];
  // d_in[3] = causal mask (handled analytically; unused)
  const float* Wq = (const float*)d_in[4];
  const float* bq = (const float*)d_in[5];
  const float* Wk = (const float*)d_in[6];
  const float* bk = (const float*)d_in[7];
  const float* Wv = (const float*)d_in[8];
  const float* bv = (const float*)d_in[9];
  const float* Wo = (const float*)d_in[10];
  const float* bo = (const float*)d_in[11];

  const size_t nElem = (size_t)B_NUM * H_NUM * S_LEN * DKD;  // 4M f16 each
  _Float16* qws = (_Float16*)d_ws;
  _Float16* kws = qws + nElem;
  _Float16* vws = kws + nElem;
  _Float16* aws = vws + nElem;  // attention output, [B, S, D] f16

  const dim3 gproj(D_DIM / 64, M_ROWS / 64);  // (16, 64)
  const float scale = 0.125f;                 // 1/sqrt(DK), folded into Q

  gemm_xwt<false, true><<<gproj, 128, 0, stream>>>(query, Wq, bq, qws, scale);
  gemm_xwt<false, true><<<gproj, 128, 0, stream>>>(key,   Wk, bk, kws, 1.0f);
  gemm_xwt<false, true><<<gproj, 128, 0, stream>>>(value, Wv, bv, vws, 1.0f);

  attn_kernel<<<dim3(S_LEN / 64, H_NUM, B_NUM), 128, 0, stream>>>(qws, kws, vws, aws);

  gemm_xwt<true, false><<<gproj, 128, 0, stream>>>(aws, Wo, bo, d_out, 1.0f);
}
